// MultiChannelSharedCrossAttention_82420422410518
// MI455X (gfx1250) — matrix operand order repaired and checked by reading, not compiled, not run on hardware
//
#include <hip/hip_runtime.h>
#include <hip/hip_bf16.h>
#include <stdint.h>

// ---------------------------------------------------------------------------
// MultiChannelSharedCrossAttention, algebraically reduced:
//   softmax over 1 key == 1  =>  attention output == V-projection
//   => f(x) = x @ (WO @ WV)^T + (WO @ bV + bO)
//   enhanced_A = w00*A + w01*f(B) + w02*f(C)   (w = row-softmax(fusion_weights))
//   enhanced_B = w10*B + w11*f(A) + w12*f(C)
//   enhanced_C = w20*C + w21*f(A) + w22*f(B)
// Main cost: 3 GEMMs [16384x1024]x[1024x1024] with split-bf16 (hi+lo)
// V_WMMA_F32_16X16X32_BF16 (3 products -> ~fp32 accuracy), double-buffered
// LDS staging; B tiles staged with GLOBAL_LOAD_ASYNC_TO_LDS when available.
// ---------------------------------------------------------------------------

typedef __attribute__((ext_vector_type(16))) __bf16 v16bf;
typedef __attribute__((ext_vector_type(8)))  float  v8f;
typedef int v4i_gcc __attribute__((vector_size(16)));   // matches builtin's pointee

#define D_MODEL 1024
#define BATCH   16384

#define BM 128   // block tile M
#define BN 128   // block tile N
#define KS 32    // k-step (bf16 WMMA K)
// 8 waves: 2 (M) x 4 (N); wave tile 64M x 32N = 4x2 subtiles of 16x16

#if defined(__AMDGCN__) && \
    __has_builtin(__builtin_amdgcn_global_load_async_to_lds_b128) && \
    __has_builtin(__builtin_amdgcn_s_wait_asynccnt)
#define USE_ASYNC_LDS 1
#else
#define USE_ASYNC_LDS 0
#endif

#if USE_ASYNC_LDS
__device__ __forceinline__ void async_cp_b128(const void* g, void* l) {
  // Builtin signature (from hipcc diagnostic): (v4i AS1*, v4i AS3*, imm, imm).
  // Global AS1 pointers share the generic bit pattern -> via-integer cast is
  // exact; LDS needs a real addrspacecast (generic aperture -> 32-bit offset).
  __builtin_amdgcn_global_load_async_to_lds_b128(
      (__attribute__((address_space(1))) v4i_gcc*)(size_t)(g),
      (__attribute__((address_space(3))) v4i_gcc*)(l), 0, 0);
}
#endif

__device__ __forceinline__ void split_bf16(float x, unsigned short& h, unsigned short& l) {
  union { float f; unsigned u; } a; a.f = x;
  unsigned short hb = (unsigned short)((a.u + 0x7FFFu + ((a.u >> 16) & 1u)) >> 16);
  union { unsigned u; float f; } hf; hf.u = ((unsigned)hb) << 16;
  float r = x - hf.f;                       // exact residual
  union { float f; unsigned u; } b; b.f = r;
  unsigned short lb = (unsigned short)((b.u + 0x7FFFu + ((b.u >> 16) & 1u)) >> 16);
  h = hb; l = lb;
}

struct ARegs { float4 v[4]; };   // 16 fp32 of one X row-segment
struct BRegs { uint4 h[2], l[2]; };

// F[m][n] = sum_k X[m][k] * Wt[k][n] + bias[n]
// Wt supplied as pre-split bf16 hi/lo planes in [k][n] layout (n contiguous).
// blockIdx.z selects among X0/X1/X2 (batched over the 3 feature matrices).
__global__ __launch_bounds__(256)
void gemm_bf16x3(const float* __restrict__ X0, const float* __restrict__ X1,
                 const float* __restrict__ X2,
                 const unsigned short* __restrict__ Bhg,
                 const unsigned short* __restrict__ Blg,
                 const float* __restrict__ bias,
                 float* __restrict__ Out, int M, int N, int K)
{
  __shared__ unsigned short Ah[2][BM * KS];
  __shared__ unsigned short Al[2][BM * KS];
  __shared__ unsigned short Bh[2][KS * BN];
  __shared__ unsigned short Bl[2][KS * BN];

  const int z = blockIdx.z;
  const float* X = (z == 0) ? X0 : ((z == 1) ? X1 : X2);
  float* O = Out + (size_t)z * (size_t)M * (size_t)N;

  const int m0 = blockIdx.x * BM;
  const int n0 = blockIdx.y * BN;
  const int t    = threadIdx.x;
  const int lane = t & 31;
  const int wid  = t >> 5;
  const int wm   = wid >> 2;   // 0..1
  const int wn   = wid & 3;    // 0..3

  // A staging geometry: 2 threads per row, 16 fp32 each
  const int ar_row = t >> 1, ar_half = t & 1;
  const float* Arowbase = X + (size_t)(m0 + ar_row) * K + ar_half * 16;
  // B staging geometry: 8 threads per k-row, 16 ushorts (2 x b128) each
  const int br_row = t >> 3, br_seg = t & 7;
  const size_t Bgbase = (size_t)br_row * N + n0 + br_seg * 16;
  const int Bloff = br_row * BN + br_seg * 16;

  v8f acc[4][2] = {};

  ARegs ar[2];
  #pragma unroll
  for (int j = 0; j < 4; j++) ar[0].v[j] = ((const float4*)Arowbase)[j];
#if USE_ASYNC_LDS
  {
    async_cp_b128(Bhg + Bgbase,     &Bh[0][Bloff]);
    async_cp_b128(Bhg + Bgbase + 8, &Bh[0][Bloff + 8]);
    async_cp_b128(Blg + Bgbase,     &Bl[0][Bloff]);
    async_cp_b128(Blg + Bgbase + 8, &Bl[0][Bloff + 8]);
  }
#else
  BRegs br[2];
  br[0].h[0] = ((const uint4*)(Bhg + Bgbase))[0];
  br[0].h[1] = ((const uint4*)(Bhg + Bgbase))[1];
  br[0].l[0] = ((const uint4*)(Blg + Bgbase))[0];
  br[0].l[1] = ((const uint4*)(Blg + Bgbase))[1];
#endif

  int cur = 0;
  for (int k0 = 0; k0 < K; k0 += KS) {
    const bool hasNext = (k0 + KS) < K;
    const int nxt = cur ^ 1;

    // prefetch next A (and B in fallback mode) into the other register set
    if (hasNext) {
      const float* np = Arowbase + (k0 + KS);
      #pragma unroll
      for (int j = 0; j < 4; j++) ar[nxt].v[j] = ((const float4*)np)[j];
#if !USE_ASYNC_LDS
      const size_t ng = Bgbase + (size_t)(k0 + KS) * N;
      br[nxt].h[0] = ((const uint4*)(Bhg + ng))[0];
      br[nxt].h[1] = ((const uint4*)(Bhg + ng))[1];
      br[nxt].l[0] = ((const uint4*)(Blg + ng))[0];
      br[nxt].l[1] = ((const uint4*)(Blg + ng))[1];
#endif
    }

    // convert + store current A into LDS buffer `cur`
    {
      unsigned* AhU = (unsigned*)Ah[cur];
      unsigned* AlU = (unsigned*)Al[cur];
      const int ubase = ar_row * (KS / 2) + ar_half * 8;
      #pragma unroll
      for (int j = 0; j < 4; j++) {
        float4 v = ar[cur].v[j];
        unsigned short h0,h1,h2,h3,l0,l1,l2,l3;
        split_bf16(v.x, h0, l0); split_bf16(v.y, h1, l1);
        split_bf16(v.z, h2, l2); split_bf16(v.w, h3, l3);
        AhU[ubase + j*2]     = (unsigned)h0 | ((unsigned)h1 << 16);
        AhU[ubase + j*2 + 1] = (unsigned)h2 | ((unsigned)h3 << 16);
        AlU[ubase + j*2]     = (unsigned)l0 | ((unsigned)l1 << 16);
        AlU[ubase + j*2 + 1] = (unsigned)l2 | ((unsigned)l3 << 16);
      }
    }
#if !USE_ASYNC_LDS
    {
      uint4* dh = (uint4*)(Bh[cur] + Bloff);
      uint4* dl = (uint4*)(Bl[cur] + Bloff);
      dh[0] = br[cur].h[0]; dh[1] = br[cur].h[1];
      dl[0] = br[cur].l[0]; dl[1] = br[cur].l[1];
    }
#endif

#if USE_ASYNC_LDS
    // B(cur) async copies (issued last iteration / prologue) must have landed
    __builtin_amdgcn_s_wait_asynccnt(0);
#endif
    __syncthreads();   // one barrier per k-step (double buffered)

#if USE_ASYNC_LDS
    // issue async B for the NEXT step after the barrier: every reader of
    // buffer `nxt` (iteration k0-KS) is ordered before this barrier.
    if (hasNext) {
      const size_t ng = Bgbase + (size_t)(k0 + KS) * N;
      async_cp_b128(Bhg + ng,     &Bh[nxt][Bloff]);
      async_cp_b128(Bhg + ng + 8, &Bh[nxt][Bloff + 8]);
      async_cp_b128(Blg + ng,     &Bl[nxt][Bloff]);
      async_cp_b128(Blg + ng + 8, &Bl[nxt][Bloff + 8]);
    }
#endif

    // ---- load fragments per the documented 16x16x32 bf16 VGPR layouts ----
    union Frag { uint4 q[2]; v16bf v; };
    Frag aH[4], aL[4], bH[2], bL[2];
    const int c0 = (lane < 16) ? 0 : 8; // A: lanes 0-15 hold K{0-7,16-23}, 16-31 hold K{8-15,24-31}
    #pragma unroll
    for (int mt = 0; mt < 4; mt++) {
      const int row = wm * 64 + mt * 16 + (lane & 15);
      const unsigned short* p = Ah[cur] + row * KS + c0;
      aH[mt].q[0] = *(const uint4*)p;
      aH[mt].q[1] = *(const uint4*)(p + 16);
      const unsigned short* q = Al[cur] + row * KS + c0;
      aL[mt].q[0] = *(const uint4*)q;
      aL[mt].q[1] = *(const uint4*)(q + 16);
    }
    #pragma unroll
    for (int nt = 0; nt < 2; nt++) {    // B: lane L holds n=L&15, K{0-7,16-23} (L<16) or K{8-15,24-31}
      const int col = wn * 32 + nt * 16 + (lane & 15);
      #pragma unroll
      for (int s = 0; s < 16; s++) {
        const int kk = (s < 8) ? (c0 + s) : (16 + c0 + (s - 8));
        bH[nt].v[s] = __builtin_bit_cast(__bf16, Bh[cur][kk * BN + col]);
        bL[nt].v[s] = __builtin_bit_cast(__bf16, Bl[cur][kk * BN + col]);
      }
    }

    // ---- 24 WMMAs: acc += Ahi*Bhi + Ahi*Blo + Alo*Bhi ----
    #pragma unroll
    for (int mt = 0; mt < 4; mt++) {
      #pragma unroll
      for (int nt = 0; nt < 2; nt++) {
        acc[mt][nt] = __builtin_amdgcn_wmma_f32_16x16x32_bf16(
            false, aH[mt].v, false, bH[nt].v, (short)0, acc[mt][nt], false, false);
        acc[mt][nt] = __builtin_amdgcn_wmma_f32_16x16x32_bf16(
            false, aH[mt].v, false, bL[nt].v, (short)0, acc[mt][nt], false, false);
        acc[mt][nt] = __builtin_amdgcn_wmma_f32_16x16x32_bf16(
            false, aL[mt].v, false, bH[nt].v, (short)0, acc[mt][nt], false, false);
      }
    }
    cur = nxt;
  }

  // ---- epilogue: C/D layout -> VGPR i is row (i + 8*(lane>=16)), col = lane&15 ----
  #pragma unroll
  for (int nt = 0; nt < 2; nt++) {
    const int col = n0 + wn * 32 + nt * 16 + (lane & 15);
    const float bv = bias ? bias[col] : 0.0f;
    #pragma unroll
    for (int mt = 0; mt < 4; mt++) {
      const int rowb = m0 + wm * 64 + mt * 16 + ((lane >> 4) << 3);
      #pragma unroll
      for (int i = 0; i < 8; i++)
        O[(size_t)(rowb + i) * N + col] = acc[mt][nt][i] + bv;
    }
  }
}

// Split fp32 matrix into bf16 hi/lo planes; optional transpose ([r][c] -> [c][r]).
__global__ __launch_bounds__(256)
void split_kernel(const float* __restrict__ W, unsigned short* __restrict__ H,
                  unsigned short* __restrict__ L, int R, int C, int transpose)
{
  const int idx = blockIdx.x * blockDim.x + threadIdx.x;
  if (idx >= R * C) return;
  const int r = idx / C, c = idx % C;
  unsigned short h, l;
  split_bf16(W[idx], h, l);
  const int o = transpose ? (c * R + r) : idx;
  H[o] = h; L[o] = l;
}

// bc[o] = WO[o,:] . bV + bO[o]   (1024 dots of length 1024 -- negligible)
__global__ __launch_bounds__(256)
void bias_kernel(const float* __restrict__ WO, const float* __restrict__ bV,
                 const float* __restrict__ bO, float* __restrict__ bc, int Dm)
{
  const int o = blockIdx.x * blockDim.x + threadIdx.x;
  if (o >= Dm) return;
  float s = bO[o];
  for (int k = 0; k < Dm; k++) s += WO[(size_t)o * Dm + k] * bV[k];
  bc[o] = s;
}

// In-place fusion: d_out currently holds [f(A), f(B), f(C)]; overwrite with
// enhanced_{A,B,C} using row-softmax(fusion_weights). Each element touched by
// exactly one thread (read-then-write) -> deterministic.
__global__ __launch_bounds__(256)
void fusion_kernel(const float* __restrict__ fA, const float* __restrict__ fB,
                   const float* __restrict__ fC, const float* __restrict__ fw,
                   float* __restrict__ out, size_t planeElems)
{
  __shared__ float wsh[9];
  if (threadIdx.x < 9) wsh[threadIdx.x] = fw[threadIdx.x];
  __syncthreads();
  float W[3][3];
  #pragma unroll
  for (int r = 0; r < 3; r++) {
    const float m  = fmaxf(wsh[r*3], fmaxf(wsh[r*3+1], wsh[r*3+2]));
    const float e0 = __expf(wsh[r*3+0] - m);
    const float e1 = __expf(wsh[r*3+1] - m);
    const float e2 = __expf(wsh[r*3+2] - m);
    const float inv = 1.0f / (e0 + e1 + e2);
    W[r][0] = e0 * inv; W[r][1] = e1 * inv; W[r][2] = e2 * inv;
  }
  const size_t i4 = (size_t)blockIdx.x * blockDim.x + threadIdx.x;
  if (i4 * 4 >= planeElems) return;
  const float4* A4 = (const float4*)fA;
  const float4* B4 = (const float4*)fB;
  const float4* C4 = (const float4*)fC;
  float4* O0 = (float4*)out;
  float4* O1 = (float4*)(out + planeElems);
  float4* O2 = (float4*)(out + 2 * planeElems);
  const float4 a = A4[i4], b = B4[i4], c = C4[i4];
  const float4 f0 = O0[i4], f1 = O1[i4], f2 = O2[i4];   // f(A), f(B), f(C)
  float4 r0, r1, r2;
#define FUSE(comp) \
  r0.comp = W[0][0]*a.comp + W[0][1]*f1.comp + W[0][2]*f2.comp; \
  r1.comp = W[1][0]*b.comp + W[1][1]*f0.comp + W[1][2]*f2.comp; \
  r2.comp = W[2][0]*c.comp + W[2][1]*f0.comp + W[2][2]*f1.comp;
  FUSE(x) FUSE(y) FUSE(z) FUSE(w)
#undef FUSE
  O0[i4] = r0; O1[i4] = r1; O2[i4] = r2;
}

extern "C" void kernel_launch(void* const* d_in, const int* in_sizes, int n_in,
                              void* d_out, int out_size, void* d_ws, size_t ws_size,
                              hipStream_t stream)
{
  (void)in_sizes; (void)n_in; (void)out_size; (void)ws_size;
  const float* fA   = (const float*)d_in[0];
  const float* fB   = (const float*)d_in[1];
  const float* fC   = (const float*)d_in[2];
  // d_in[3..6] = WQ_w, WQ_b, WK_w, WK_b: dead (softmax over one key == 1)
  const float* WV_w = (const float*)d_in[7];
  const float* WV_b = (const float*)d_in[8];
  const float* WO_w = (const float*)d_in[9];
  const float* WO_b = (const float*)d_in[10];
  const float* fw   = (const float*)d_in[11];
  float* out = (float*)d_out;
  char*  ws  = (char*)d_ws;

  const int D = D_MODEL;
  // workspace layout (~12 MB):
  unsigned short* WVh  = (unsigned short*)(ws + (size_t)0);          // 2 MB  [k][i]
  unsigned short* WVl  = (unsigned short*)(ws + ((size_t)2 << 20));  // 2 MB
  float*          Wc   = (float*)         (ws + ((size_t)4 << 20));  // 4 MB  [o][i]
  unsigned short* WcTh = (unsigned short*)(ws + ((size_t)8 << 20));  // 2 MB  [i][o]
  unsigned short* WcTl = (unsigned short*)(ws + ((size_t)10 << 20)); // 2 MB
  float*          bc   = (float*)         (ws + ((size_t)12 << 20)); // 4 KB

  const int total = D * D;
  // 1) split WV (already [k][i], no transpose) for the combine GEMM's B operand
  split_kernel<<<(total + 255) / 256, 256, 0, stream>>>(WV_w, WVh, WVl, D, D, 0);
  // 2) Wc = WO @ WV  (1024^3, same WMMA kernel, z-batch of 1, no bias)
  dim3 g1(D / BM, D / BN, 1);
  gemm_bf16x3<<<g1, 256, 0, stream>>>(WO_w, WO_w, WO_w, WVh, WVl, nullptr, Wc, D, D, D);
  // 3) split + transpose Wc -> [k=i][n=o] bf16 planes for the main GEMMs
  split_kernel<<<(total + 255) / 256, 256, 0, stream>>>(Wc, WcTh, WcTl, D, D, 1);
  // 4) bc = WO @ bV + bO
  bias_kernel<<<(D + 255) / 256, 256, 0, stream>>>(WO_w, WV_b, WO_b, bc, D);
  // 5) F(A), F(B), F(C) -> d_out planes 0,1,2 (z-batched GEMM)
  dim3 g2(BATCH / BM, D / BN, 3);
  gemm_bf16x3<<<g2, 256, 0, stream>>>(fA, fB, fC, WcTh, WcTl, bc, out, BATCH, D, D);
  // 6) in-place fusion with softmaxed fusion weights
  const size_t plane = (size_t)BATCH * D;
  const unsigned nblk = (unsigned)((plane / 4 + 255) / 256);
  fusion_kernel<<<nblk, 256, 0, stream>>>(fA, fB, fC, fw, out, plane);
}
